// PositionAttentionLayer_20057497272358
// MI455X (gfx1250) — matrix operand order repaired and checked by reading, not compiled, not run on hardware
//
#include <hip/hip_runtime.h>

typedef __attribute__((ext_vector_type(16))) _Float16 v16h;
typedef __attribute__((ext_vector_type(8)))  _Float16 v8h;
typedef __attribute__((ext_vector_type(8)))  float    v8f;
typedef __attribute__((ext_vector_type(4)))  float    v4f;

#define WMMA_F16(a, b, c) \
  __builtin_amdgcn_wmma_f32_16x16x32_f16(false, (a), false, (b), (short)0, (c), false, false)

static constexpr int Bn   = 8;
static constexpr int Cch  = 512;
static constexpr int Nn   = 4096;   // H*W = 64*64
static constexpr int CQK  = 80;
static constexpr int CQKP = 96;     // pad 80 -> 96 so K-dim = 3 x 32
static constexpr int KP   = 104;    // padded K stride (halves) for q/k tiles, 16B aligned
static constexpr int VP   = 40;     // padded j stride (halves) for v tile / x tile

// ---- fragment builders (ISA 7.12.2 16-bit layouts) -------------------------
// A 16x32: lane row = lane&15; elements 0..7 <-> K = ks*32 + 8*hi + 0..7,
//          elements 8..15 <-> K = ks*32 + 16 + 8*hi + 0..7
__device__ __forceinline__ v16h frag_a(const _Float16* row, int ks, int hi) {
  const v8h lo  = *(const v8h*)(row + ks * 32 + 8 * hi);
  const v8h hiv = *(const v8h*)(row + ks * 32 + 16 + 8 * hi);
  v16h a;
  #pragma unroll
  for (int i = 0; i < 8; ++i) { a[i] = lo[i]; a[i + 8] = hiv[i]; }
  return a;
}
// B 32x16: lane col = lane&15; elements 0..7 <-> K = ks*32 + 8*hi + 0..7,
//          elements 8..15 <-> K = ks*32 + 16 + 8*hi + 0..7
__device__ __forceinline__ v16h frag_b(const _Float16* row, int ks, int hi) {
  const v8h lo  = *(const v8h*)(row + ks * 32 + 8 * hi);
  const v8h hiv = *(const v8h*)(row + ks * 32 + 16 + 8 * hi);
  v16h b;
  #pragma unroll
  for (int i = 0; i < 8; ++i) { b[i] = lo[i]; b[i + 8] = hiv[i]; }
  return b;
}

// -------- Projection: out[b,m,n] = relu(scale[m] * sum_c W[m,c]*x[b,c,n] + bias[m])
// Block = 8 waves sharing one LDS-staged x tile [32k x 64n]; wave w computes the
// 16-row tile ot = og*8 + w over a 64-column strip.
__global__ void __launch_bounds__(256)
proj_gemm_kernel(const float* __restrict__ x,      // [B][C][N] fp32
                 const float* __restrict__ W,      // [Ovalid][C] fp32
                 const float* __restrict__ scale,  // [Ovalid]
                 const float* __restrict__ bias,   // [Ovalid]
                 _Float16* __restrict__ out,       // [B][Opad][N] f16
                 int Ovalid, int Opad, int OG)
{
  __shared__ alignas(16) _Float16 xs[64 * VP];   // x tile, K-contiguous, 5 KB

  const int tid  = threadIdx.x;
  const int wid  = tid >> 5;
  const int lane = tid & 31;
  const int r    = lane & 15;
  const int hi   = lane >> 4;

  const int nTilesN = Nn / 64;
  int nt  = blockIdx.x % nTilesN;
  int tmp = blockIdx.x / nTilesN;
  int og  = tmp % OG;
  int b   = tmp / OG;

  const int nTilesO = Opad / 16;
  const int ot = og * 8 + wid;
  const bool active  = ot < nTilesO;
  const int m0 = ot * 16;
  const bool compute = active && (m0 < Ovalid);
  const int n0 = nt * 64;

  const float* xB = x + (size_t)b * Cch * Nn;
  _Float16* outB  = out + (size_t)b * Opad * Nn;

  v8f acc[4] = {{}, {}, {}, {}};

  for (int k0 = 0; k0 < Cch; k0 += 32) {
    // stage + convert x tile: 32 k-rows x 64 cols, stored transposed [n][k]
    for (int ch = tid; ch < 32 * 16; ch += 256) {
      int k = ch >> 4, nq = ch & 15;
      v4f xv = *(const v4f*)(xB + (size_t)(k0 + k) * Nn + n0 + nq * 4);
      #pragma unroll
      for (int i = 0; i < 4; ++i) xs[(nq * 4 + i) * VP + k] = (_Float16)xv[i];
    }
    __syncthreads();

    if (compute) {
      const float* wrow = W + (size_t)(m0 + r) * Cch + k0;
      v4f w0 = *(const v4f*)(wrow + 8 * hi);
      v4f w1 = *(const v4f*)(wrow + 8 * hi + 4);
      v4f w2 = *(const v4f*)(wrow + 16 + 8 * hi);
      v4f w3 = *(const v4f*)(wrow + 16 + 8 * hi + 4);
      v16h a;
      #pragma unroll
      for (int i = 0; i < 4; ++i) {
        a[i]      = (_Float16)w0[i];
        a[i + 4]  = (_Float16)w1[i];
        a[i + 8]  = (_Float16)w2[i];
        a[i + 12] = (_Float16)w3[i];
      }
      // hoist all B fragments so ds_loads overlap the WMMA chain
      v16h bf[4];
      #pragma unroll
      for (int st = 0; st < 4; ++st)
        bf[st] = frag_b(&xs[(st * 16 + r) * VP], 0, hi);
      #pragma unroll
      for (int st = 0; st < 4; ++st)
        acc[st] = WMMA_F16(a, bf[st], acc[st]);
    }
    __syncthreads();
  }

  if (!active) return;
  if (compute) {
    #pragma unroll
    for (int st = 0; st < 4; ++st)
      #pragma unroll
      for (int v = 0; v < 8; ++v) {
        int m = m0 + v + 8 * hi;
        float y = acc[st][v] * scale[m] + bias[m];
        y = y > 0.0f ? y : 0.0f;
        outB[(size_t)m * Nn + n0 + st * 16 + r] = (_Float16)y;
      }
  } else {  // zero padding rows (80..95 of q/k)
    #pragma unroll
    for (int st = 0; st < 4; ++st)
      #pragma unroll
      for (int v = 0; v < 8; ++v)
        outB[(size_t)(m0 + v + 8 * hi) * Nn + n0 + st * 16 + r] = (_Float16)0.0f;
  }
}

// -------- Flash attention + epilogue: out = gamma * (softmax(q^T k) v^T)^T + x
// One block per (batch, 16-row query tile); j streamed in tiles of 32.
__global__ void __launch_bounds__(256)
attn_kernel(const _Float16* __restrict__ qh,   // [B][96][N]
            const _Float16* __restrict__ kh,   // [B][96][N]
            const _Float16* __restrict__ vh,   // [B][512][N]
            const float* __restrict__ x,       // [B][C][N]
            const float* __restrict__ gamma,   // [C][N]
            float* __restrict__ out)           // [B][C][N]
{
  __shared__ alignas(16) _Float16 qs[16 * KP];    // q tile [i][K]   3.3 KB
  __shared__ alignas(16) _Float16 ksm[32 * KP];   // k tile [j][K]   6.7 KB
  __shared__ alignas(16) _Float16 vsm[Cch * VP];  // v tile [c][j]  41.0 KB
  __shared__ alignas(16) float    ss[16 * 32];    // S tile [i][j]   2 KB
  __shared__ alignas(16) _Float16 ps[16 * 32];    // P tile [i][j]   1 KB
  __shared__ float mrow[16], lrow[16], arow[16];

  const int tid  = threadIdx.x;
  const int wid  = tid >> 5;
  const int lane = tid & 31;
  const int r    = lane & 15;
  const int hi   = lane >> 4;

  const int it = blockIdx.x % (Nn / 16);
  const int b  = blockIdx.x / (Nn / 16);
  const int i0 = it * 16;

  const _Float16* qB = qh + (size_t)b * CQKP * Nn;
  const _Float16* kB = kh + (size_t)b * CQKP * Nn;
  const _Float16* vB = vh + (size_t)b * Cch * Nn;

  // stage q tile transposed: qs[i][k]
  for (int idx = tid; idx < CQKP * 16; idx += 256) {
    int k = idx >> 4, i = idx & 15;
    qs[i * KP + k] = qB[(size_t)k * Nn + i0 + i];
  }
  if (tid < 16) { mrow[tid] = -1e30f; lrow[tid] = 0.0f; }
  __syncthreads();

  // q A-fragments (K=96 -> 3 k-steps), registers for all j-steps
  v16h qa[3];
  #pragma unroll
  for (int ks3 = 0; ks3 < 3; ++ks3)
    qa[ks3] = frag_a(&qs[r * KP], ks3, hi);

  v8f oacc[4] = {{}, {}, {}, {}};  // each wave owns 64 channels: 4 x 16

  for (int j0 = 0; j0 < Nn; j0 += 32) {
    // stage k tile transposed: ksm[j][k]  (vector global load, scalar LDS scatter)
    for (int k = tid; k < CQKP; k += 256) {
      const v8h* src = (const v8h*)(kB + (size_t)k * Nn + j0);
      v8h r0 = src[0], r1 = src[1], r2 = src[2], r3 = src[3];
      #pragma unroll
      for (int j = 0; j < 8; ++j) {
        ksm[(j +  0) * KP + k] = r0[j];
        ksm[(j +  8) * KP + k] = r1[j];
        ksm[(j + 16) * KP + k] = r2[j];
        ksm[(j + 24) * KP + k] = r3[j];
      }
    }
    // stage v tile: vsm[c][j], fully vectorized both sides
    for (int c = tid; c < Cch; c += 256) {
      const v8h* src = (const v8h*)(vB + (size_t)c * Nn + j0);
      v8h* dst = (v8h*)&vsm[c * VP];
      dst[0] = src[0]; dst[1] = src[1]; dst[2] = src[2]; dst[3] = src[3];
    }
    // prefetch next j-tile of v (global_prefetch_b8 path)
    if (j0 + 32 < Nn) {
      __builtin_prefetch((const void*)(vB + (size_t)(tid * 2) * Nn + j0 + 32), 0, 1);
      __builtin_prefetch((const void*)(vB + (size_t)(tid * 2 + 1) * Nn + j0 + 32), 0, 1);
    }
    __syncthreads();

    // S = q_i^T k_j : 16x32, waves 0 (j 0..15) and 1 (j 16..31)
    if (wid < 2) {
      v16h bf[3];
      #pragma unroll
      for (int ks3 = 0; ks3 < 3; ++ks3)
        bf[ks3] = frag_b(&ksm[(wid * 16 + r) * KP], ks3, hi);
      v8f s = {};
      #pragma unroll
      for (int ks3 = 0; ks3 < 3; ++ks3)
        s = WMMA_F16(qa[ks3], bf[ks3], s);
      #pragma unroll
      for (int v = 0; v < 8; ++v)
        ss[(v + 8 * hi) * 32 + wid * 16 + r] = s[v];
    }
    __syncthreads();

    // online softmax bookkeeping in fp32 (one thread per query row)
    if (tid < 16) {
      float mold = mrow[tid];
      float mt = mold;
      #pragma unroll 8
      for (int j = 0; j < 32; ++j) mt = fmaxf(mt, ss[tid * 32 + j]);
      float alpha = __expf(mold - mt);
      float lsum = 0.0f;
      #pragma unroll 8
      for (int j = 0; j < 32; ++j) {
        float p = __expf(ss[tid * 32 + j] - mt);
        ps[tid * 32 + j] = (_Float16)p;
        lsum += p;
      }
      mrow[tid] = mt;
      lrow[tid] = lrow[tid] * alpha + lsum;
      arow[tid] = alpha;
    }
    __syncthreads();

    float av[8];
    #pragma unroll
    for (int v = 0; v < 8; ++v) av[v] = arow[v + 8 * hi];

    v16h pa = frag_a(&ps[r * 32], 0, hi);   // P A-fragment (16 x 32)

    // O[i,c] = O[i,c]*alpha + P @ v^T  (all 8 waves, 4 channel tiles each)
    // hoist all 4 v fragments so ds_loads overlap the WMMA chain
    v16h vb[4];
    #pragma unroll
    for (int t = 0; t < 4; ++t)
      vb[t] = frag_b(&vsm[(size_t)((wid * 4 + t) * 16 + r) * VP], 0, hi);
    #pragma unroll
    for (int t = 0; t < 4; ++t) {
      #pragma unroll
      for (int v = 0; v < 8; ++v) oacc[t][v] *= av[v];
      oacc[t] = WMMA_F16(pa, vb[t], oacc[t]);
    }
    __syncthreads();
  }

  float linv[8];
  #pragma unroll
  for (int v = 0; v < 8; ++v) linv[v] = 1.0f / lrow[v + 8 * hi];

  const float* xB = x + (size_t)b * Cch * Nn;
  float* outB = out + (size_t)b * Cch * Nn;
  #pragma unroll
  for (int t = 0; t < 4; ++t) {
    int c = (wid * 4 + t) * 16 + r;
    #pragma unroll
    for (int v = 0; v < 8; ++v) {
      int ig = i0 + v + 8 * hi;
      size_t idx = (size_t)c * Nn + ig;
      outB[idx] = gamma[idx] * (oacc[t][v] * linv[v]) + xB[idx];
    }
  }
}

extern "C" void kernel_launch(void* const* d_in, const int* in_sizes, int n_in,
                              void* d_out, int out_size, void* d_ws, size_t ws_size,
                              hipStream_t stream) {
  const float* x     = (const float*)d_in[0];
  const float* Wq    = (const float*)d_in[1];
  const float* Wk    = (const float*)d_in[2];
  const float* Wv    = (const float*)d_in[3];
  const float* sq    = (const float*)d_in[4];
  const float* bq    = (const float*)d_in[5];
  const float* sk    = (const float*)d_in[6];
  const float* bk    = (const float*)d_in[7];
  const float* sv    = (const float*)d_in[8];
  const float* bv    = (const float*)d_in[9];
  const float* gamma = (const float*)d_in[10];
  float* out = (float*)d_out;

  _Float16* qh = (_Float16*)d_ws;                       // [B][96][N]  6 MB
  _Float16* kh = qh + (size_t)Bn * CQKP * Nn;           // [B][96][N]  6 MB
  _Float16* vh = kh + (size_t)Bn * CQKP * Nn;           // [B][512][N] 32 MB

  dim3 blk(256);
  const int OGqk = 1;                                   // ceil(96/128)
  const int OGv  = (Cch / 16) / 8;                      // 4
  int qkBlocks = Bn * (Nn / 64) * OGqk;                 // 512
  int vBlocks  = Bn * (Nn / 64) * OGv;                  // 2048

  proj_gemm_kernel<<<qkBlocks, blk, 0, stream>>>(x, Wq, sq, bq, qh, CQK, CQKP, OGqk);
  proj_gemm_kernel<<<qkBlocks, blk, 0, stream>>>(x, Wk, sk, bk, kh, CQK, CQKP, OGqk);
  proj_gemm_kernel<<<vBlocks,  blk, 0, stream>>>(x, Wv, sv, bv, vh, Cch, Cch, OGv);
  attn_kernel<<<Bn * (Nn / 16), blk, 0, stream>>>(qh, kh, vh, x, gamma, out);
}
